// DSA_884763263409
// MI455X (gfx1250) — matrix operand order repaired and checked by reading, not compiled, not run on hardware
//
#include <hip/hip_runtime.h>
#include <hip/hip_bf16.h>
#include <math.h>

// ---------------------------------------------------------------------------
// DCNv3 block (B=4, C=256, H=W=64, G=8, P=9) for MI455X / gfx1250.
//  - All 1x1 projections: f16 WMMA GEMMs (v_wmma_f32_16x16x32_f16, fp32 acc).
//  - GEMM "B" operands are f16 K-major mirrors (B,HW,256): tile staging is a
//    raw 2D copy driven by the Tensor Data Mover (tensor_load_to_lds), now
//    software-pipelined: next tile's TDM DMA + A staging overlap the current
//    tile's WMMAs; s_wait_tensorcnt fences only at the buffer-swap barrier.
//  - Workspace: 4 f32 planes + offset/mask + 3 f16 mirror slots ~= 101.5 MiB
//    (fits the 192 MB L2; HBM sees ~first-touch x + final out only).
// ---------------------------------------------------------------------------

typedef __attribute__((ext_vector_type(16))) _Float16 v16h;
typedef __attribute__((ext_vector_type(8)))  _Float16 v8h;
typedef __attribute__((ext_vector_type(2)))  _Float16 v2h;
typedef __attribute__((ext_vector_type(8)))  float    v8f;
typedef __attribute__((ext_vector_type(4)))  float    v4f;
typedef __attribute__((ext_vector_type(4)))  unsigned u32x4;
typedef __attribute__((ext_vector_type(8)))  int      i32x8;
typedef __attribute__((ext_vector_type(4)))  int      i32x4;

#if __has_builtin(__builtin_amdgcn_tensor_load_to_lds) && \
    __has_builtin(__builtin_amdgcn_s_wait_tensorcnt)
#define DSA_HAVE_TDM 1
#else
#define DSA_HAVE_TDM 0
#endif

namespace {
constexpr int Bn = 4, Cc = 256, Hh = 64, Ww = 64;
constexpr int Gg = 8, Pp = 9;
constexpr int HW = Hh * Ww;     // 4096
constexpr int Kd = 256;         // GEMM reduction dim, always 256
constexpr int Nd = HW;          // GEMM N per batch
}

__device__ __forceinline__ float gelu_f(float v) {
  return 0.5f * v * (1.0f + erff(v * 0.7071067811865475f));  // exact GELU
}

// ---------------------------------------------------------------------------
// GEMM: Out[b,m,n] = act( sum_k W[k,m] * X16[b,n,k] + bias[m] ) (+resid)
//   W    : Kd x MTOT f32 (reference layout, transposed+converted in staging)
//   X16  : (Bn, Nd, Kd) f16 K-major mirror (raw-copy tile -> TDM)
//   OutF : (Bn, MTOT, Nd) f32 NCHW         [if WF32]
//   Out16: (Bn, Nd, MTOT) f16 K-major      [if WF16]
// Block 256 thr = 8 waves, tile 64x128, K-step 32, wave strip 16x64.
// Double-buffered LDS: TDM(B) + VALU(A) staging of tile k+1 overlap WMMA of k.
// ---------------------------------------------------------------------------
template <int MTOT, int ACT, bool RESID, bool WF32, bool WF16>
__global__ __launch_bounds__(256) void dsa_gemm(
    const float* __restrict__ Wmat,
    const _Float16* __restrict__ X16,
    const float* __restrict__ bias,
    float* __restrict__ OutF,
    const float* __restrict__ resid,
    _Float16* __restrict__ Out16)
{
  __shared__ _Float16 As[2][64 * 32];    // [m][k], ping-pong
  __shared__ _Float16 Bs[2][128 * 32];   // [n][k], ping-pong

  const int tid  = threadIdx.x;
  const int lane = tid & 31;
  const int wave = tid >> 5;
  const int wm   = wave >> 1;            // 0..3
  const int wn   = wave & 1;             // 0..1
  const int m0   = blockIdx.y * 64;
  const int n0   = blockIdx.x * 128;
  const int bb   = blockIdx.z;
  const _Float16* X16b = X16 + (size_t)bb * Nd * Kd;

  v8f acc[4] = {};

  // A staging map: 4 m x 2 k per thread (vector loads, packed b32 LDS stores)
  const int amm = (tid & 15) * 4;        // 0..60
  const int akk = (tid >> 4) * 2;        // 0..30
  // B staging map (fallback path): one 8-half chunk per thread
  const int bn  = tid >> 1;              // 0..127
  const int bkh = (tid & 1) * 8;         // 0 or 8

  // ---- stagers -----------------------------------------------------------
  auto stage_a = [&](int k0, _Float16* dstA) {
    bool aval = true;
    if constexpr (MTOT % 64 != 0) aval = (m0 + amm < MTOT);
    v4f r0 = {}, r1 = {};
    if (aval) {
      const float* s0 = Wmat + (size_t)(k0 + akk) * MTOT + (m0 + amm);
      r0 = *(const v4f*)s0;
      r1 = *(const v4f*)(s0 + MTOT);
    }
    #pragma unroll
    for (int i = 0; i < 4; ++i) {
      v2h p;
      p[0] = (_Float16)r0[i];
      p[1] = (_Float16)r1[i];
      *(v2h*)(dstA + (amm + i) * 32 + akk) = p;
    }
  };

#if DSA_HAVE_TDM
  auto tdm_issue = [&](int k0, _Float16* dstB) {   // wave 0 only
    const unsigned lds_b = (unsigned)(size_t)dstB;
    const unsigned long long ga =
        (unsigned long long)(const void*)(X16b + (size_t)n0 * Kd + k0);
    u32x4 g0;
    g0[0] = 1u;                                        // count=1, user D#
    g0[1] = lds_b;                                     // lds_addr
    g0[2] = (unsigned)(ga & 0xFFFFFFFFu);              // global_addr lo
    g0[3] = (unsigned)((ga >> 32) & 0x01FFFFFFu) | (2u << 30);  // type=2
    i32x8 g1;
    g1[0] = (int)(1u << 16);                 // data_size=2B, no multicast
    g1[1] = (int)(256u << 16);               // tensor_dim0=256 (lo16)
    g1[2] = (int)(((unsigned)(Bn * HW) & 0xFFFFu) << 16);        // dim1 lo16
    g1[3] = (int)((((unsigned)(Bn * HW) >> 16) & 0xFFFFu) | (32u << 16)); // tile_dim0=32
    g1[4] = (int)128u;                       // tile_dim1=128, tile_dim2=0
    g1[5] = (int)256u;                       // tensor_dim0_stride lo32
    g1[6] = (int)((256u & 0xFFFFu) << 16);   // stride0 hi=0, stride1 lo16
    g1[7] = 0;
    i32x4 gz; gz[0] = 0; gz[1] = 0; gz[2] = 0; gz[3] = 0;
#if defined(__clang_major__) && __clang_major__ >= 23
    i32x8 gz8; gz8[0]=0; gz8[1]=0; gz8[2]=0; gz8[3]=0; gz8[4]=0; gz8[5]=0; gz8[6]=0; gz8[7]=0;
    __builtin_amdgcn_tensor_load_to_lds(g0, g1, gz, gz, gz8, 0);
#else
    __builtin_amdgcn_tensor_load_to_lds(g0, g1, gz, gz, 0);
#endif
  };
#else
  auto stage_b_manual = [&](int k0, _Float16* dstB) {
    const v8h bv = *(const v8h*)(X16b + (size_t)(n0 + bn) * Kd + k0 + bkh);
    *(v8h*)(dstB + bn * 32 + bkh) = bv;
  };
#endif

  // ---- prologue: tile 0 ---------------------------------------------------
#if DSA_HAVE_TDM
  if (wave == 0) tdm_issue(0, &Bs[0][0]);
#else
  stage_b_manual(0, &Bs[0][0]);
#endif
  stage_a(0, &As[0][0]);
#if DSA_HAVE_TDM
  if (wave == 0) __builtin_amdgcn_s_wait_tensorcnt(0);
#endif
  __syncthreads();

  // ---- pipelined main loop ------------------------------------------------
  constexpr int NK = Kd / 32;            // 8
  #pragma unroll
  for (int kk = 0; kk < NK; ++kk) {
    const int cur = kk & 1;
    const int nxt = cur ^ 1;

    if (kk + 1 < NK) {                   // launch tile k+1 while computing k
#if DSA_HAVE_TDM
      if (wave == 0) tdm_issue((kk + 1) * 32, &Bs[nxt][0]);
#else
      stage_b_manual((kk + 1) * 32, &Bs[nxt][0]);
#endif
      stage_a((kk + 1) * 32, &As[nxt][0]);
    }

    // fragments + WMMA from the current buffers
    const int ml = wm * 16 + (lane & 15);
    const int ka = (lane >> 4) * 8;      // A k-groups 0..7/16..23 | 8..15/24..31
    const v8h a_lo = *(const v8h*)(&As[cur][0] + ml * 32 + ka);
    const v8h a_hi = *(const v8h*)(&As[cur][0] + ml * 32 + ka + 16);
    const v16h afrag = __builtin_shufflevector(
        a_lo, a_hi, 0, 1, 2, 3, 4, 5, 6, 7, 8, 9, 10, 11, 12, 13, 14, 15);

    const int kb = (lane >> 4) * 8;      // B k-groups 0..7/16..23 | 8..15/24..31
    #pragma unroll
    for (int j = 0; j < 4; ++j) {
      const int nl = wn * 64 + j * 16 + (lane & 15);
      const v8h b_lo = *(const v8h*)(&Bs[cur][0] + nl * 32 + kb);
      const v8h b_hi = *(const v8h*)(&Bs[cur][0] + nl * 32 + kb + 16);
      const v16h bfrag = __builtin_shufflevector(
          b_lo, b_hi, 0, 1, 2, 3, 4, 5, 6, 7, 8, 9, 10, 11, 12, 13, 14, 15);
      acc[j] = __builtin_amdgcn_wmma_f32_16x16x32_f16(
          false, afrag, false, bfrag, (short)0, acc[j], false, false);
    }

    if (kk + 1 < NK) {                   // fence DMA, swap buffers
#if DSA_HAVE_TDM
      if (wave == 0) __builtin_amdgcn_s_wait_tensorcnt(0);
#endif
      __syncthreads();
    }
  }

  // ---- epilogue: lane holds n=(lane&15); VGPR r -> m = base + r ----------
  const int mrow = m0 + wm * 16 + (lane >> 4) * 8;   // 8 consecutive m
  #pragma unroll
  for (int j = 0; j < 4; ++j) {
    const int n = n0 + wn * 64 + j * 16 + (lane & 15);
    float vv[8];
    #pragma unroll
    for (int r = 0; r < 8; ++r) {
      bool mval = true;
      if constexpr (MTOT % 64 != 0) mval = (mrow + r < MTOT);
      float v = acc[j][r] + (mval ? bias[mrow + r] : 0.f);
      if constexpr (ACT == 1) v = gelu_f(v);
      if constexpr (WF32) {
        if (mval) {
          const size_t o = ((size_t)bb * MTOT + (mrow + r)) * Nd + n;
          if constexpr (RESID) v += resid[o];
          OutF[o] = v;
        }
      }
      vv[r] = v;
    }
    if constexpr (WF16) {
      v8h hv;
      #pragma unroll
      for (int r = 0; r < 8; ++r) hv[r] = (_Float16)vv[r];
      *(v8h*)(Out16 + ((size_t)bb * Nd + n) * MTOT + mrow) = hv;
    }
  }
}

// ---------------------------------------------------------------------------
// NCHW f32 -> (B,HW,C) f16 K-major mirror
// ---------------------------------------------------------------------------
__global__ void dsa_cvt_mirror(const float* __restrict__ in,
                               _Float16* __restrict__ out16)
{
  const int idx = blockIdx.x * blockDim.x + threadIdx.x;  // B*HW
  if (idx >= Bn * HW) return;
  const int b = idx / HW, hw = idx % HW;
  const float* src = in + (size_t)b * Cc * HW + hw;
  _Float16* dst = out16 + ((size_t)b * HW + hw) * Cc;
  #pragma unroll
  for (int c8 = 0; c8 < Cc / 8; ++c8) {
    v8h h;
    #pragma unroll
    for (int i = 0; i < 8; ++i) h[i] = (_Float16)src[(size_t)(c8 * 8 + i) * HW];
    *(v8h*)(dst + c8 * 8) = h;
  }
}

// ---------------------------------------------------------------------------
// Depthwise KSxKS conv, NCHW, zero pad; optional f16 mirror output
// ---------------------------------------------------------------------------
template <int KS, int PADK, bool MIR>
__global__ void dsa_dwconv(const float* __restrict__ in,
                           const float* __restrict__ w,
                           const float* __restrict__ bias,
                           float* __restrict__ out,
                           _Float16* __restrict__ mir16)
{
  const int idx = blockIdx.x * blockDim.x + threadIdx.x;  // B*C*HW
  if (idx >= Bn * Cc * HW) return;
  const int hw = idx % HW;
  const int bc = idx / HW;
  const int c  = bc % Cc;
  const int b  = bc / Cc;
  const int x  = hw % Ww, y = hw / Ww;
  float s = bias[c];
  #pragma unroll
  for (int i = 0; i < KS; ++i) {
    const int yy = y + i - PADK;
    if (yy < 0 || yy >= Hh) continue;
    #pragma unroll
    for (int j = 0; j < KS; ++j) {
      const int xx = x + j - PADK;
      if (xx < 0 || xx >= Ww) continue;
      s += w[(c * KS + i) * KS + j] * in[(size_t)bc * HW + yy * Ww + xx];
    }
  }
  out[idx] = s;
  if constexpr (MIR)
    mir16[((size_t)b * HW + hw) * Cc + c] = (_Float16)s;
}

// ---------------------------------------------------------------------------
// LayerNorm over C (NCHW strided) + exact GELU -> f16 K-major mirror only
// ---------------------------------------------------------------------------
__global__ void dsa_ln_gelu(const float* __restrict__ xin,
                            const float* __restrict__ g,
                            const float* __restrict__ bvec,
                            _Float16* __restrict__ out16)
{
  const int idx = blockIdx.x * blockDim.x + threadIdx.x;  // B*HW
  if (idx >= Bn * HW) return;
  const int b = idx / HW, hw = idx % HW;
  const float* base = xin + (size_t)b * Cc * HW + hw;
  float mu = 0.f;
  for (int c = 0; c < Cc; ++c) mu += base[(size_t)c * HW];
  mu *= (1.0f / Cc);
  float var = 0.f;
  for (int c = 0; c < Cc; ++c) {
    const float d = base[(size_t)c * HW] - mu;
    var += d * d;
  }
  var *= (1.0f / Cc);
  const float rs = rsqrtf(var + 1e-5f);
  _Float16* dst = out16 + ((size_t)b * HW + hw) * Cc;
  for (int c8 = 0; c8 < Cc / 8; ++c8) {
    v8h h;
    #pragma unroll
    for (int i = 0; i < 8; ++i) {
      const int c = c8 * 8 + i;
      const float v = (base[(size_t)c * HW] - mu) * rs * g[c] + bvec[c];
      h[i] = (_Float16)gelu_f(v);
    }
    *(v8h*)(dst + c8 * 8) = h;
  }
}

// ---------------------------------------------------------------------------
// In-place softmax over P=9 points per group (mask: B x 72 x HW)
// ---------------------------------------------------------------------------
__global__ void dsa_softmax9(float* __restrict__ mask)
{
  const int idx = blockIdx.x * blockDim.x + threadIdx.x;  // B*G*HW
  if (idx >= Bn * Gg * HW) return;
  const int hw = idx % HW;
  const int t  = idx / HW;
  const int g  = t % Gg, b = t / Gg;
  float* mp = mask + ((size_t)b * (Gg * Pp) + g * Pp) * HW + hw;
  float mx = -3.4e38f;
  #pragma unroll
  for (int p = 0; p < Pp; ++p) mx = fmaxf(mx, mp[(size_t)p * HW]);
  float e[Pp], s = 0.f;
  #pragma unroll
  for (int p = 0; p < Pp; ++p) { e[p] = __expf(mp[(size_t)p * HW] - mx); s += e[p]; }
  const float inv = 1.0f / s;
  #pragma unroll
  for (int p = 0; p < Pp; ++p) mp[(size_t)p * HW] = e[p] * inv;
}

// ---------------------------------------------------------------------------
// DCNv3 core: one thread per (b,hw,g); coords amortized over 32 channels.
// PAD=1,STRIDE=DIL=1 => fx = w+gx+off_x, fy = h+gy+off_y on unpadded image.
// Validity folded into the bilinear weights (branch-free clamped gathers).
// ---------------------------------------------------------------------------
__global__ __launch_bounds__(256) void dsa_dcn_sample(
    const float* __restrict__ xproj,   // B,C,HW f32
    const float* __restrict__ off,     // B,144,HW
    const float* __restrict__ mask,    // B,72,HW (softmaxed)
    _Float16* __restrict__ out16)      // B,HW,C f16
{
  const int idx = blockIdx.x * blockDim.x + threadIdx.x;  // B*G*HW, hw fastest
  if (idx >= Bn * Gg * HW) return;
  const int hw = idx % HW;
  const int t  = idx / HW;
  const int g  = t % Gg, b = t / Gg;
  const int x = hw % Ww, y = hw / Ww;
  const float* plane0 = xproj + ((size_t)b * Cc + g * 32) * HW;
  const float* op = off  + (size_t)b * (Gg * Pp * 2) * HW + hw;
  const float* mp = mask + (size_t)b * (Gg * Pp) * HW + hw;

  float acc[32];
  #pragma unroll
  for (int cc = 0; cc < 32; ++cc) acc[cc] = 0.f;

  for (int p = 0; p < Pp; ++p) {
    const float gx = (float)(p / 3 - 1);   // torch meshgrid 'ij', x outer
    const float gy = (float)(p % 3 - 1);
    const float ox = op[(size_t)(g * (Pp * 2) + p * 2 + 0) * HW];
    const float oy = op[(size_t)(g * (Pp * 2) + p * 2 + 1) * HW];
    const float m  = mp[(size_t)(g * Pp + p) * HW];
    const float fx = (float)x + gx + ox;
    const float fy = (float)y + gy + oy;
    const float x0f = floorf(fx), y0f = floorf(fy);
    const float ax = fx - x0f, ay = fy - y0f;
    const int x0 = (int)x0f, y0 = (int)y0f;
    const int x1i = x0 + 1, y1i = y0 + 1;
    const bool vx0 = (x0 >= 0) && (x0 < Ww), vx1 = (x1i >= 0) && (x1i < Ww);
    const bool vy0 = (y0 >= 0) && (y0 < Hh), vy1 = (y1i >= 0) && (y1i < Hh);
    const float w00 = (vx0 && vy0) ? (1.f - ax) * (1.f - ay) * m : 0.f;
    const float w01 = (vx1 && vy0) ? ax * (1.f - ay) * m : 0.f;
    const float w10 = (vx0 && vy1) ? (1.f - ax) * ay * m : 0.f;
    const float w11 = (vx1 && vy1) ? ax * ay * m : 0.f;
    const int cx0 = min(max(x0, 0), Ww - 1), cx1 = min(max(x1i, 0), Ww - 1);
    const int cy0 = min(max(y0, 0), Hh - 1), cy1 = min(max(y1i, 0), Hh - 1);
    const int i00 = cy0 * Ww + cx0, i01 = cy0 * Ww + cx1;
    const int i10 = cy1 * Ww + cx0, i11 = cy1 * Ww + cx1;
    #pragma unroll
    for (int cc = 0; cc < 32; ++cc) {
      const float* pl = plane0 + (size_t)cc * HW;
      acc[cc] += pl[i00] * w00 + pl[i01] * w01 + pl[i10] * w10 + pl[i11] * w11;
    }
  }
  _Float16* o = out16 + ((size_t)b * HW + hw) * Cc + g * 32;
  #pragma unroll
  for (int c8 = 0; c8 < 4; ++c8) {
    v8h h;
    #pragma unroll
    for (int i = 0; i < 8; ++i) h[i] = (_Float16)acc[c8 * 8 + i];
    *(v8h*)(o + c8 * 8) = h;
  }
}

// ---------------------------------------------------------------------------
// m16 = f16( u * attnc )  -> (B,HW,C) mirror
// ---------------------------------------------------------------------------
__global__ void dsa_mul16(const float* __restrict__ a,
                          const float* __restrict__ bsrc,
                          _Float16* __restrict__ o16)
{
  const int idx = blockIdx.x * blockDim.x + threadIdx.x;  // B*C*HW
  if (idx >= Bn * Cc * HW) return;
  const int hw = idx % HW;
  const int bc = idx / HW;
  const int c  = bc % Cc;
  const int b  = bc / Cc;
  o16[((size_t)b * HW + hw) * Cc + c] = (_Float16)(a[idx] * bsrc[idx]);
}

// ---------------------------------------------------------------------------
extern "C" void kernel_launch(void* const* d_in, const int* in_sizes, int n_in,
                              void* d_out, int out_size, void* d_ws, size_t ws_size,
                              hipStream_t stream) {
  (void)in_sizes; (void)n_in; (void)out_size; (void)ws_size;
  const float* x         = (const float*)d_in[0];
  const float* proj1_w   = (const float*)d_in[1];
  const float* proj1_b   = (const float*)d_in[2];
  const float* conv0_w   = (const float*)d_in[3];
  const float* conv0_b   = (const float*)d_in[4];
  const float* inproj_w  = (const float*)d_in[5];
  const float* inproj_b  = (const float*)d_in[6];
  const float* dw_w      = (const float*)d_in[7];
  const float* dw_b      = (const float*)d_in[8];
  const float* ln_g      = (const float*)d_in[9];
  const float* ln_bb     = (const float*)d_in[10];
  const float* off_w     = (const float*)d_in[11];
  const float* off_b     = (const float*)d_in[12];
  const float* mask_w    = (const float*)d_in[13];
  const float* mask_b    = (const float*)d_in[14];
  const float* outproj_w = (const float*)d_in[15];
  const float* outproj_b = (const float*)d_in[16];
  const float* conv_w    = (const float*)d_in[17];
  const float* conv_b    = (const float*)d_in[18];
  const float* proj2_w   = (const float*)d_in[19];
  const float* proj2_b   = (const float*)d_in[20];
  float* out = (float*)d_out;

  const size_t PLANE = (size_t)Bn * Cc * HW;   // 4,194,304 elements
  float* ws     = (float*)d_ws;
  float* buf_h  = ws;                 // u = gelu(proj1) — live to the end
  float* buf_hc = ws + PLANE;         // dw5 out (f32, feeds dw3)
  float* buf_xp = ws + 2 * PLANE;     // xproj -> later attnc
  float* buf_x1 = ws + 3 * PLANE;     // dw3 raw (LN reads it)
  float* buf_of = ws + 4 * PLANE;                             // B,144,HW
  float* buf_mk = buf_of + (size_t)Bn * (Gg * Pp * 2) * HW;   // B,72,HW
  _Float16* h16 = (_Float16*)(buf_mk + (size_t)Bn * (Gg * Pp) * HW);
  _Float16* slotA = h16;              // x16  -> attn16
  _Float16* slotB = h16 + PLANE;      // hc16 -> dcn16
  _Float16* slotC = h16 + 2 * PLANE;  // x1_16 -> m16

  const dim3 blk(256);
  const int NELT = Bn * Cc * HW;

  // 0) x -> f16 K-major mirror
  dsa_cvt_mirror<<<(Bn * HW) / 256, blk, 0, stream>>>(x, slotA);
  // 1) h = gelu(x @ proj1_w + b)
  dsa_gemm<256, 1, false, true, false><<<dim3(Nd / 128, 4, Bn), blk, 0, stream>>>(
      proj1_w, slotA, proj1_b, buf_h, nullptr, nullptr);
  // 2) hc = dwconv5(h) (+ f16 mirror for inproj)
  dsa_dwconv<5, 2, true><<<NELT / 256, blk, 0, stream>>>(buf_h, conv0_w, conv0_b, buf_hc, slotB);
  // 3) xproj = hc @ inproj_w + b
  dsa_gemm<256, 0, false, true, false><<<dim3(Nd / 128, 4, Bn), blk, 0, stream>>>(
      inproj_w, slotB, inproj_b, buf_xp, nullptr, nullptr);
  // 4) x1raw = dwconv3(hc)
  dsa_dwconv<3, 1, false><<<NELT / 256, blk, 0, stream>>>(buf_hc, dw_w, dw_b, buf_x1, nullptr);
  // 5) x1 = gelu(ln(x1raw)) -> f16 mirror
  dsa_ln_gelu<<<(Bn * HW) / 256, blk, 0, stream>>>(buf_x1, ln_g, ln_bb, slotC);
  // 6) offsets = x1 @ off_w + b   (M=144)
  dsa_gemm<144, 0, false, true, false><<<dim3(Nd / 128, 3, Bn), blk, 0, stream>>>(
      off_w, slotC, off_b, buf_of, nullptr, nullptr);
  // 7) mask = x1 @ mask_w + b     (M=72)
  dsa_gemm<72, 0, false, true, false><<<dim3(Nd / 128, 2, Bn), blk, 0, stream>>>(
      mask_w, slotC, mask_b, buf_mk, nullptr, nullptr);
  // 8) softmax over the 9 points / group
  dsa_softmax9<<<(Bn * Gg * HW) / 256, blk, 0, stream>>>(buf_mk);
  // 9) deformable sampling -> dcn16 (slotB; hc16 dead)
  dsa_dcn_sample<<<(Bn * Gg * HW) / 256, blk, 0, stream>>>(buf_xp, buf_of, buf_mk, slotB);
  // 10) attn16 = dcn @ outproj_w + b (f16 mirror only; slotA free)
  dsa_gemm<256, 0, false, false, true><<<dim3(Nd / 128, 4, Bn), blk, 0, stream>>>(
      outproj_w, slotB, outproj_b, nullptr, nullptr, slotA);
  // 11) attnc = attn @ conv_w + b -> buf_xp (xproj dead)
  dsa_gemm<256, 0, false, true, false><<<dim3(Nd / 128, 4, Bn), blk, 0, stream>>>(
      conv_w, slotA, conv_b, buf_xp, nullptr, nullptr);
  // 12) m16 = u * attnc (slotC free)
  dsa_mul16<<<NELT / 256, blk, 0, stream>>>(buf_h, buf_xp, slotC);
  // 13) out = m @ proj2_w + b + shortcut(x)
  dsa_gemm<256, 0, true, true, false><<<dim3(Nd / 128, 4, Bn), blk, 0, stream>>>(
      proj2_w, slotC, proj2_b, out, x, nullptr);
}
